// MultiHeadAttention_32633161515354
// MI455X (gfx1250) — matrix operand order repaired and checked by reading, not compiled, not run on hardware
//
#include <hip/hip_runtime.h>
#include <hip/hip_bf16.h>
#include <math.h>

#define DEVFN __device__ __forceinline__

typedef __attribute__((ext_vector_type(16))) __bf16 v16bf;
typedef __attribute__((ext_vector_type(8)))  float  v8f;

union Frag {
  v16bf v;
  uint4 u[2];
};

constexpr int Bsz = 4, S = 2048, Hdim = 1024, NHh = 16, HDd = 64;
constexpr int Mtot = Bsz * S;  // 8192

DEVFN unsigned short f2bf_bits(float f) {
  __bf16 h = (__bf16)f;
  return __builtin_bit_cast(unsigned short, h);
}

DEVFN v8f wmma_bf16(const Frag& a, const Frag& b, v8f c) {
  // D = A(16x32 bf16) * B(32x16 bf16) + C(16x16 f32)
  return __builtin_amdgcn_wmma_f32_16x16x32_bf16(false, a.v, false, b.v,
                                                 (short)0, c, false, false);
}

// ---------------------------------------------------------------------------
// fp32 -> bf16 conversion (grid-stride)
// ---------------------------------------------------------------------------
__global__ void f32_to_bf16_kernel(const float* __restrict__ in,
                                   unsigned short* __restrict__ out, int n) {
  int i = blockIdx.x * blockDim.x + threadIdx.x;
  int stride = gridDim.x * blockDim.x;
  for (; i < n; i += stride) out[i] = f2bf_bits(in[i]);
}

// ---------------------------------------------------------------------------
// WMMA GEMM:  Y[m,n] = sum_k A[m,k] * W[n,k] + bias[n]
//   A: (Mtot x Hdim) bf16 row-major, W: (Hdim x Hdim) bf16 row-major (N x K)
// MODE 0: bf16 out, layout (b, nh, s, hd)      (for Q and K)
// MODE 2: bf16 out, layout (b, nh, hd, s)      (for V transposed)
// MODE 3: f32 out, row-major (m, Hdim)         (final output projection)
// Block: 256 threads = 8 waves; wave tile 16(M) x 64(N); block tile 64 x 128.
// ---------------------------------------------------------------------------
template <int MODE>
__global__ __launch_bounds__(256) void gemm_bf16_kernel(
    const unsigned short* __restrict__ A, const unsigned short* __restrict__ W,
    const float* __restrict__ bias, void* __restrict__ out) {
  const int lane = threadIdx.x & 31;
  const int wv = threadIdx.x >> 5;
  const int hf = lane >> 4;  // half-wave
  const int ln = lane & 15;
  const int wm = wv & 3, wn = wv >> 2;  // wm 0..3, wn 0..1
  const int m0 = blockIdx.y * 64 + wm * 16;
  const int n0 = blockIdx.x * 128 + wn * 64;
  const int K = Hdim;

  v8f c[4] = {};
  const unsigned short* arow = A + (size_t)(m0 + ln) * K;

  for (int kk = 0; kk < K; kk += 32) {
    Frag a;  // A 16x32: lane row = ln, VGPR0-3 <-> k = hf*8+0..7, VGPR4-7 <-> 16+hf*8+0..7
    a.u[0] = *(const uint4*)(arow + kk + hf * 8);
    a.u[1] = *(const uint4*)(arow + kk + 16 + hf * 8);
#pragma unroll
    for (int t = 0; t < 4; ++t) {
      Frag b;  // B 32x16: lane col = ln, VGPR0-3 <-> k = hf*8+0..7, VGPR4-7 <-> 16+hf*8+0..7
      const unsigned short* wp =
          W + (size_t)(n0 + t * 16 + ln) * K + kk + hf * 8;
      b.u[0] = *(const uint4*)(wp);
      b.u[1] = *(const uint4*)(wp + 16);
      c[t] = wmma_bf16(a, b, c[t]);
    }
  }

#pragma unroll
  for (int t = 0; t < 4; ++t) {
    const int n = n0 + t * 16 + ln;
    const float bval = bias[n];
    if constexpr (MODE == 3) {
      float* o = (float*)out;
#pragma unroll
      for (int r = 0; r < 8; ++r) {
        const int mr = m0 + r + 8 * hf;  // C layout: lane holds rows r+8*hf, col ln
        o[(size_t)mr * Hdim + n] = c[t][r] + bval;
      }
    } else {
      unsigned short* o = (unsigned short*)out;
      const int nh = n >> 6, hd = n & 63;
#pragma unroll
      for (int r = 0; r < 8; ++r) {
        const int mr = m0 + r + 8 * hf;
        const int bb = mr >> 11, ss = mr & 2047;
        size_t idx;
        if constexpr (MODE == 2)
          idx = ((size_t)(bb * NHh + nh) * HDd + hd) * S + ss;
        else
          idx = ((size_t)(bb * NHh + nh) * S + ss) * HDd + hd;
        o[idx] = f2bf_bits(c[t][r] + bval);
      }
    }
  }
}

// ---------------------------------------------------------------------------
// Flash attention (causal). Block = 128 threads = 4 waves.
// Wave handles 16 query rows; iterates 32-key blocks with online softmax.
// Q,K layout: (head, s, hd) bf16 ; V layout: (head, hd, s) bf16 (transposed)
// Output: (b, s, H) bf16
// ---------------------------------------------------------------------------
__global__ __launch_bounds__(128) void flash_attn_kernel(
    const unsigned short* __restrict__ Q, const unsigned short* __restrict__ Kt,
    const unsigned short* __restrict__ Vt, unsigned short* __restrict__ O) {
  __shared__ unsigned short plds[4][16 * 32];  // per-wave P staging (16x32 bf16)
  const int lane = threadIdx.x & 31;
  const int wv = threadIdx.x >> 5;
  const int hf = lane >> 4, ln = lane & 15;
  const int head = blockIdx.y;
  const int q0 = blockIdx.x * 64 + wv * 16;
  const unsigned short* Qh = Q + (size_t)head * S * HDd;
  const unsigned short* Kh = Kt + (size_t)head * S * HDd;
  const unsigned short* Vh = Vt + (size_t)head * HDd * S;

  // Q fragments, d 0..31 and 32..63 (loaded once)
  Frag aq0, aq1;
  {
    const unsigned short* qp = Qh + (size_t)(q0 + ln) * HDd;
    aq0.u[0] = *(const uint4*)(qp + hf * 8);
    aq0.u[1] = *(const uint4*)(qp + 16 + hf * 8);
    aq1.u[0] = *(const uint4*)(qp + 32 + hf * 8);
    aq1.u[1] = *(const uint4*)(qp + 48 + hf * 8);
  }

  v8f acc[4] = {};
  float rowmax[8], rowsum[8];
#pragma unroll
  for (int r = 0; r < 8; ++r) {
    rowmax[r] = -__builtin_inff();
    rowsum[r] = 0.f;
  }

  int kend = (q0 + 16 + 31) & ~31;  // keys <= q0+15, rounded to 32
  if (kend > S) kend = S;
  unsigned short* myp = plds[wv];

  for (int kb = 0; kb < kend; kb += 32) {
    if (kb + 32 < kend) {  // prefetch next K/V block (global_prefetch_b8)
      __builtin_prefetch(Kh + (size_t)(kb + 32 + ln) * HDd, 0, 0);
      __builtin_prefetch(Vh + (size_t)ln * S + kb + 32, 0, 0);
    }

    // ---- scores: two 16x16 tiles over keys [kb,kb+16) and [kb+16,kb+32)
    v8f c0 = {}, c1 = {};
    {
      Frag bk;
      const unsigned short* kp = Kh + (size_t)(kb + ln) * HDd + hf * 8;
      bk.u[0] = *(const uint4*)kp;
      bk.u[1] = *(const uint4*)(kp + 16);
      c0 = wmma_bf16(aq0, bk, c0);
      bk.u[0] = *(const uint4*)(kp + 32);
      bk.u[1] = *(const uint4*)(kp + 48);
      c0 = wmma_bf16(aq1, bk, c0);
      kp = Kh + (size_t)(kb + 16 + ln) * HDd + hf * 8;
      bk.u[0] = *(const uint4*)kp;
      bk.u[1] = *(const uint4*)(kp + 16);
      c1 = wmma_bf16(aq0, bk, c1);
      bk.u[0] = *(const uint4*)(kp + 32);
      bk.u[1] = *(const uint4*)(kp + 48);
      c1 = wmma_bf16(aq1, bk, c1);
    }

    // ---- scale, causal mask, online softmax
    const bool needmask = (kb + 31 > q0);
#pragma unroll
    for (int r = 0; r < 8; ++r) {
      float s0 = c0[r] * 0.125f;  // 1/sqrt(64)
      float s1 = c1[r] * 0.125f;
      if (needmask) {
        const int q = q0 + r + 8 * hf;
        if (kb + ln > q) s0 = -__builtin_inff();
        if (kb + 16 + ln > q) s1 = -__builtin_inff();
      }
      float t = fmaxf(s0, s1);
      t = fmaxf(t, __shfl_xor(t, 1));
      t = fmaxf(t, __shfl_xor(t, 2));
      t = fmaxf(t, __shfl_xor(t, 4));
      t = fmaxf(t, __shfl_xor(t, 8));
      const float nm = fmaxf(rowmax[r], t);
      const float alpha = __expf(rowmax[r] - nm);
      rowmax[r] = nm;
      const float p0 = __expf(s0 - nm);
      const float p1 = __expf(s1 - nm);
      float ps = p0 + p1;
      ps += __shfl_xor(ps, 1);
      ps += __shfl_xor(ps, 2);
      ps += __shfl_xor(ps, 4);
      ps += __shfl_xor(ps, 8);
      rowsum[r] = rowsum[r] * alpha + ps;
#pragma unroll
      for (int t2 = 0; t2 < 4; ++t2) acc[t2][r] *= alpha;
      // stage P (C layout -> row-major LDS), 16x32 bf16
      const int mrow = r + 8 * hf;
      myp[mrow * 32 + ln] = f2bf_bits(p0);
      myp[mrow * 32 + 16 + ln] = f2bf_bits(p1);
    }

    // ---- read P back in A-fragment layout (same wave: DS ops are in-order)
    Frag pa;
    pa.u[0] = *(const uint4*)&myp[ln * 32 + hf * 8];
    pa.u[1] = *(const uint4*)&myp[ln * 32 + 16 + hf * 8];

    // ---- P(16x32) x V(32x64): one WMMA per 16-wide hd tile
#pragma unroll
    for (int t = 0; t < 4; ++t) {
      Frag bvf;
      const unsigned short* vp =
          Vh + (size_t)(t * 16 + ln) * S + kb + hf * 8;
      bvf.u[0] = *(const uint4*)vp;
      bvf.u[1] = *(const uint4*)(vp + 16);
      acc[t] = wmma_bf16(pa, bvf, acc[t]);
    }
  }

  // ---- normalize and store (b, s, H) bf16
  const int bb = head >> 4, nh = head & 15;
#pragma unroll
  for (int r = 0; r < 8; ++r) {
    const float inv = 1.0f / rowsum[r];
    const int mrow = q0 + r + 8 * hf;
    unsigned short* orow = O + ((size_t)(bb * S + mrow)) * Hdim + nh * HDd;
#pragma unroll
    for (int t = 0; t < 4; ++t)
      orow[t * 16 + ln] = f2bf_bits(acc[t][r] * inv);
  }
}

// ---------------------------------------------------------------------------
extern "C" void kernel_launch(void* const* d_in, const int* in_sizes, int n_in,
                              void* d_out, int out_size, void* d_ws,
                              size_t ws_size, hipStream_t stream) {
  const float* x = (const float*)d_in[0];
  const float* wq = (const float*)d_in[1];
  const float* bq = (const float*)d_in[2];
  const float* wk = (const float*)d_in[3];
  const float* bk = (const float*)d_in[4];
  const float* wvp = (const float*)d_in[5];
  const float* bv = (const float*)d_in[6];
  const float* wo = (const float*)d_in[7];
  const float* bo = (const float*)d_in[8];

  unsigned short* ws = (unsigned short*)d_ws;
  const size_t NX = (size_t)Mtot * Hdim;  // 8388608
  const size_t NW = (size_t)Hdim * Hdim;  // 1048576
  unsigned short* xbf = ws;
  unsigned short* wqbf = xbf + NX;
  unsigned short* wkbf = wqbf + NW;
  unsigned short* wvbf = wkbf + NW;
  unsigned short* wobf = wvbf + NW;
  unsigned short* Qbf = wobf + NW;
  unsigned short* Kbf = Qbf + NX;
  unsigned short* Vtb = Kbf + NX;
  unsigned short* Obf = Vtb + NX;
  // total: 5*NX + 4*NW elements = ~92.3 MB of workspace

  f32_to_bf16_kernel<<<2048, 256, 0, stream>>>(x, xbf, (int)NX);
  f32_to_bf16_kernel<<<512, 256, 0, stream>>>(wq, wqbf, (int)NW);
  f32_to_bf16_kernel<<<512, 256, 0, stream>>>(wk, wkbf, (int)NW);
  f32_to_bf16_kernel<<<512, 256, 0, stream>>>(wvp, wvbf, (int)NW);
  f32_to_bf16_kernel<<<512, 256, 0, stream>>>(wo, wobf, (int)NW);

  dim3 ggrid(Hdim / 128, Mtot / 64);  // (8, 128)
  gemm_bf16_kernel<0><<<ggrid, 256, 0, stream>>>(xbf, wqbf, bq, Qbf);
  gemm_bf16_kernel<0><<<ggrid, 256, 0, stream>>>(xbf, wkbf, bk, Kbf);
  gemm_bf16_kernel<2><<<ggrid, 256, 0, stream>>>(xbf, wvbf, bv, Vtb);

  dim3 agrid(S / 64, Bsz * NHh);  // (32, 64)
  flash_attn_kernel<<<agrid, 128, 0, stream>>>(Qbf, Kbf, Vtb, Obf);

  gemm_bf16_kernel<3><<<ggrid, 256, 0, stream>>>(Obf, wobf, bo, d_out);
}
